// Decoder_55379308315272
// MI455X (gfx1250) — matrix operand order repaired and checked by reading, not compiled, not run on hardware
//
#include <hip/hip_runtime.h>
#include <hip/hip_bf16.h>
#include <math.h>

#define BV   50000
#define BE   128
#define BH   512
#define BB   128
#define BS   400
#define H2   1024

typedef __bf16 bf16_t;
typedef __attribute__((ext_vector_type(16))) __bf16 v16bf;
typedef __attribute__((ext_vector_type(8)))  float  v8f;

#define DEV static __device__ __forceinline__

DEV float sigmoidf_(float x) { return 1.0f / (1.0f + __expf(-x)); }

// ---------------------------------------------------------------------------
// f32 LDS tiles: 128 rows x 32 cols, stride 36 dwords (144B, 16B aligned).
// Row-to-row bank advance 36 mod 64 -> 16 fragment lanes on distinct banks.
// ---------------------------------------------------------------------------
#define ASTRIDE 36

// Copy 16 contiguous floats (64B) global -> LDS via async DMA (ASYNCcnt).
// GVS mode: global addr = SGPR base + VGPR offset + offset:N,
// LDS addr  = VGPR dst + offset:N (same literal shifts both sides).
DEV void async_stage_row16(unsigned ldsAddr, const float* saddrBase, unsigned byteOff)
{
    asm volatile("global_load_async_to_lds_b128 %0, %1, %2 offset:0"
                 :: "v"(ldsAddr), "v"(byteOff), "s"(saddrBase) : "memory");
    asm volatile("global_load_async_to_lds_b128 %0, %1, %2 offset:16"
                 :: "v"(ldsAddr), "v"(byteOff), "s"(saddrBase) : "memory");
    asm volatile("global_load_async_to_lds_b128 %0, %1, %2 offset:32"
                 :: "v"(ldsAddr), "v"(byteOff), "s"(saddrBase) : "memory");
    asm volatile("global_load_async_to_lds_b128 %0, %1, %2 offset:48"
                 :: "v"(ldsAddr), "v"(byteOff), "s"(saddrBase) : "memory");
}

DEV void wait_async_all()
{
    asm volatile("s_wait_asynccnt 0" ::: "memory");
}

// Fragment builders from f32 LDS, converting to bf16 at the point of use.
// A fragment per ISA 7.12.2 (16-bit A 16x32): lane<16 row, K{0-7,16-23};
// lane>=16 same row, K{8-15,24-31}.
DEV v16bf load_afrag_f32(const float* Af, int mt, int lane)
{
    const int row = mt * 16 + (lane & 15);
    const int kb  = (lane & 16) ? 8 : 0;
    const float* p = Af + row * ASTRIDE;
    v16bf a;
#pragma unroll
    for (int t = 0; t < 8; ++t) a[t]     = (bf16_t)p[kb + t];
#pragma unroll
    for (int t = 0; t < 8; ++t) a[t + 8] = (bf16_t)p[kb + 16 + t];
    return a;
}

// B fragment: lane = column; lanes 0-15 hold K{0-7,16-23}, lanes 16-31 hold K{8-15,24-31}.
DEV v16bf load_bfrag_f32(const float* Bf, int nt, int lane)
{
    const int col = nt * 16 + (lane & 15);
    const int kb  = (lane & 16) ? 8 : 0;
    const float* p = Bf + col * ASTRIDE + kb;
    v16bf b;
#pragma unroll
    for (int t = 0; t < 16; ++t) b[t] = (bf16_t)p[(t < 8) ? t : t + 8];
    return b;
}

__global__ void kz_zero(float* p, int n)
{
    int i = blockIdx.x * 256 + threadIdx.x;
    if (i < n) p[i] = 0.0f;
}

// ---------------------------------------------------------------------------
// K1: embed + input proj + LSTM + sBias = s_hat@Ws.T + bs + bh + bc
// ---------------------------------------------------------------------------
__global__ __launch_bounds__(256)
void k1_lstm(const int* prev_target, const float* h0, const float* c0,
             const float* prev_ctx, const float* embedding,
             const float* Wxc, const float* bxc,
             const float* W_ih, const float* W_hh,
             const float* b_ih, const float* b_hh,
             const float* Ws, const float* bs, const float* bh, const float* bc,
             float* out_h, float* out_c,
             float* ws_x, float* ws_shat, float* ws_sbias)
{
    const int b = blockIdx.x, t = threadIdx.x;
    __shared__ float sh_xin[H2 + BE];
    __shared__ float sh_x[BE];
    __shared__ float sh_h0[BH];
    __shared__ float sh_shat[H2];

    const float* emb = embedding + (long)prev_target[b] * BE;
    if (t < BE) sh_xin[t] = emb[t];
    for (int i = t; i < H2; i += 256) sh_xin[BE + i] = prev_ctx[(long)b * H2 + i];
    for (int i = t; i < BH; i += 256) sh_h0[i] = h0[(long)b * BH + i];
    __syncthreads();

    if (t < BE) {
        const float* wr = Wxc + (long)t * (H2 + BE);
        float acc = bxc[t];
        for (int k = 0; k < H2 + BE; ++k) acc = fmaf(sh_xin[k], wr[k], acc);
        sh_x[t] = acc;
        ws_x[(long)b * BE + t] = acc;
    }
    __syncthreads();

    for (int rep = 0; rep < 2; ++rep) {
        const int d = t + rep * 256;
        float g4[4];
#pragma unroll
        for (int gi = 0; gi < 4; ++gi) {
            const int j = gi * BH + d;
            const float* wi = W_ih + (long)j * BE;
            const float* wh = W_hh + (long)j * BH;
            float acc = b_ih[j] + b_hh[j];
            for (int k = 0; k < BE; ++k) acc = fmaf(sh_x[k], wi[k], acc);
            for (int k = 0; k < BH; ++k) acc = fmaf(sh_h0[k], wh[k], acc);
            g4[gi] = acc;
        }
        const float cc = sigmoidf_(g4[1]) * c0[(long)b * BH + d] +
                         sigmoidf_(g4[0]) * tanhf(g4[2]);
        const float hh = sigmoidf_(g4[3]) * tanhf(cc);
        out_h[(long)b * BH + d] = hh;
        out_c[(long)b * BH + d] = cc;
        sh_shat[d]      = hh;
        sh_shat[BH + d] = cc;
        ws_shat[(long)b * H2 + d]      = hh;
        ws_shat[(long)b * H2 + BH + d] = cc;
    }
    __syncthreads();

    for (int d = t; d < H2; d += 256) {
        const float* wr = Ws + (long)d * H2;
        float acc = bs[d] + bh[d] + bc[d];
        for (int k = 0; k < H2; ++k) acc = fmaf(sh_shat[k], wr[k], acc);
        ws_sbias[(long)b * H2 + d] = acc;
    }
}

// ---------------------------------------------------------------------------
// K2: fused attention-energy GEMM, bf16 WMMA, async-to-LDS software pipeline.
//   e[row] = sum_d tanh( enc[row,:]@Wh[d,:] + sBias[b,d] + cov[row]*Wc[d] )*v[d]
// Grid (8 n-blocks, 400 m-blocks), 256 threads = 8 waves, tile 128x128, K=32.
// Single LDS buffer: fragments decouple compute from LDS, so slab k+1's DMA
// overlaps slab k's WMMA burst.
// ---------------------------------------------------------------------------
__global__ __launch_bounds__(256)
void k2_attn_gemm(const float* enc, const float* Wh, const float* sbias,
                  const float* cov, const float* Wc, const float* vvec,
                  float* ws_e)
{
    __shared__ alignas(16) float Af[128 * ASTRIDE];
    __shared__ alignas(16) float Bf[128 * ASTRIDE];
    __shared__ float e_sh[128];

    const int tid  = threadIdx.x;
    const int lane = tid & 31;
    const int wave = tid >> 5;
    const int wy = wave >> 2;
    const int wx = wave & 3;
    const long mbase = (long)blockIdx.y * 128;
    const long nbase = (long)blockIdx.x * 128;

    if (tid < 128) e_sh[tid] = 0.0f;

    // per-thread staging addresses (row si, 16-float segment sj)
    const int si = tid >> 1;
    const int sj = (tid & 1) * 16;
    const unsigned ldsA = (unsigned)(uintptr_t)(&Af[si * ASTRIDE + sj]);
    const unsigned ldsB = (unsigned)(uintptr_t)(&Bf[si * ASTRIDE + sj]);
    const float* encBase = enc + mbase * H2;
    const float* whBase  = Wh  + nbase * H2;
    const unsigned rowOff = (unsigned)((si * H2 + sj) * 4);

    v8f acc[4][2];
#pragma unroll
    for (int im = 0; im < 4; ++im)
#pragma unroll
        for (int in = 0; in < 2; ++in) acc[im][in] = (v8f){};

    // prologue: DMA slab 0, build fragments
    async_stage_row16(ldsA, encBase, rowOff);
    async_stage_row16(ldsB, whBase,  rowOff);
    wait_async_all();
    __syncthreads();

    v16bf af[4], bfr[2];
#pragma unroll
    for (int im = 0; im < 4; ++im) af[im] = load_afrag_f32(Af, wy * 4 + im, lane);
#pragma unroll
    for (int in = 0; in < 2; ++in) bfr[in] = load_bfrag_f32(Bf, wx * 2 + in, lane);

    const int NS = H2 / 32;
    for (int ks = 0; ks < NS; ++ks) {
        // all waves' fragment ds_loads drained (s_wait_dscnt 0 inside barrier),
        // LDS is free for refill
        __syncthreads();
        const bool more = (ks + 1 < NS);
        if (more) {
            const unsigned off = rowOff + (unsigned)((ks + 1) * 32 * 4);
            async_stage_row16(ldsA, encBase, off);
            async_stage_row16(ldsB, whBase,  off);
        }
        // XDL burst on current fragments while the DMA fills slab ks+1
#pragma unroll
        for (int im = 0; im < 4; ++im)
#pragma unroll
            for (int in = 0; in < 2; ++in)
                acc[im][in] = __builtin_amdgcn_wmma_f32_16x16x32_bf16(
                    false, af[im], false, bfr[in], (short)0, acc[im][in],
                    false, false);
        if (more) {
            wait_async_all();
            __syncthreads();
#pragma unroll
            for (int im = 0; im < 4; ++im) af[im] = load_afrag_f32(Af, wy * 4 + im, lane);
#pragma unroll
            for (int in = 0; in < 2; ++in) bfr[in] = load_bfrag_f32(Bf, wx * 2 + in, lane);
        }
    }

    // epilogue: tanh + dot with v, half-wave reduce, LDS combine, atomic e
#pragma unroll
    for (int im = 0; im < 4; ++im) {
        const int mt = wy * 4 + im;
#pragma unroll
        for (int r = 0; r < 8; ++r) {
            const int mloc = mt * 16 + ((lane & 16) ? r + 8 : r);
            const long row = mbase + mloc;
            const int  bb  = (int)(row / BS);
            const float covr = cov[row];
            float part = 0.0f;
#pragma unroll
            for (int in = 0; in < 2; ++in) {
                const int d = (int)nbase + (wx * 2 + in) * 16 + (lane & 15);
                const float arg = acc[im][in][r] + sbias[(long)bb * H2 + d] +
                                  covr * Wc[d];
                part += tanhf(arg) * vvec[d];
            }
            part += __shfl_xor(part, 1, 32);
            part += __shfl_xor(part, 2, 32);
            part += __shfl_xor(part, 4, 32);
            part += __shfl_xor(part, 8, 32);
            if ((lane & 15) == 0) atomicAdd(&e_sh[mloc], part);
        }
    }
    __syncthreads();
    if (tid < 128) atomicAdd(&ws_e[mbase + tid], e_sh[tid]);
}

// ---------------------------------------------------------------------------
// K3: softmax(e) -> attn; coverage = prev_cov + attn
// ---------------------------------------------------------------------------
__global__ __launch_bounds__(512)
void k3_softmax_e(const float* ws_e, const float* prev_cov,
                  float* attn_out, float* cov_out)
{
    const int b = blockIdx.x, t = threadIdx.x;
    __shared__ float red[512];
    const float e = (t < BS) ? ws_e[(long)b * BS + t] : -__builtin_inff();
    red[t] = e; __syncthreads();
    for (int s = 256; s > 0; s >>= 1) {
        if (t < s) red[t] = fmaxf(red[t], red[t + s]);
        __syncthreads();
    }
    const float m = red[0]; __syncthreads();
    const float ex = (t < BS) ? __expf(e - m) : 0.0f;
    red[t] = ex; __syncthreads();
    for (int s = 256; s > 0; s >>= 1) {
        if (t < s) red[t] += red[t + s];
        __syncthreads();
    }
    const float inv = 1.0f / red[0];
    if (t < BS) {
        const float a = ex * inv;
        attn_out[(long)b * BS + t] = a;
        cov_out[(long)b * BS + t]  = prev_cov[(long)b * BS + t] + a;
    }
}

// ---------------------------------------------------------------------------
// K4: context = attn @ enc
// ---------------------------------------------------------------------------
__global__ __launch_bounds__(256)
void k4_context(const float* attn, const float* enc, float* ctx_out)
{
    const int b = blockIdx.y;
    const int d = blockIdx.x * 256 + threadIdx.x;
    __shared__ float sa[BS];
    for (int i = threadIdx.x; i < BS; i += 256) sa[i] = attn[(long)b * BS + i];
    __syncthreads();
    float acc = 0.0f;
    const float* ep = enc + ((long)b * BS) * H2 + d;
    for (int s = 0; s < BS; ++s) acc = fmaf(sa[s], ep[(long)s * H2], acc);
    ctx_out[(long)b * H2 + d] = acc;
}

// ---------------------------------------------------------------------------
// K5a: t1 = [h, context] @ Wv1.T + bv1
// ---------------------------------------------------------------------------
__global__ __launch_bounds__(256)
void k5a_t1(const float* out_h, const float* ctx, const float* Wv1,
            const float* bv1, float* ws_t1)
{
    const int b = blockIdx.x, t = threadIdx.x;
    __shared__ float hc[BH + H2];
    for (int i = t; i < BH; i += 256) hc[i] = out_h[(long)b * BH + i];
    for (int i = t; i < H2; i += 256) hc[BH + i] = ctx[(long)b * H2 + i];
    __syncthreads();
    for (int j = t; j < BH; j += 256) {
        const float* wr = Wv1 + (long)j * (BH + H2);
        float acc = bv1[j];
        for (int k = 0; k < BH + H2; ++k) acc = fmaf(hc[k], wr[k], acc);
        ws_t1[(long)b * BH + j] = acc;
    }
}

// ---------------------------------------------------------------------------
// K5c: p_gen
// ---------------------------------------------------------------------------
__global__ __launch_bounds__(128)
void k5c_pgen(const float* ws_shat, const float* ctx, const float* ws_x,
              const float* Wpg, const float* bpg, float* out_pg)
{
    const int b = threadIdx.x;
    float acc = bpg[0];
    const float* w = Wpg;
    for (int k = 0; k < H2; ++k) acc = fmaf(ws_shat[(long)b * H2 + k], w[k], acc);
    for (int k = 0; k < H2; ++k) acc = fmaf(ctx[(long)b * H2 + k], w[H2 + k], acc);
    for (int k = 0; k < BE; ++k) acc = fmaf(ws_x[(long)b * BE + k], w[2 * H2 + k], acc);
    out_pg[b] = sigmoidf_(acc);
}

// ---------------------------------------------------------------------------
// K5b: logits = t1 @ Wv2.T + bv2 -> final_dist rows (stride Vext).
// Same async-pipelined bf16 WMMA core; tail rows (n >= 50000) are simply not
// copied (EXEC-masked async issue) and their output columns never stored.
// ---------------------------------------------------------------------------
__global__ __launch_bounds__(256)
void k5b_vocab_gemm(const float* ws_t1, const float* Wv2, const float* bv2,
                    float* outF, int Vext)
{
    __shared__ alignas(16) float Af[128 * ASTRIDE];
    __shared__ alignas(16) float Bf[128 * ASTRIDE];

    const int tid  = threadIdx.x;
    const int lane = tid & 31;
    const int wave = tid >> 5;
    const int wy = wave >> 2;
    const int wx = wave & 3;
    const long nbase = (long)blockIdx.x * 128;
    long nv = (long)BV - nbase;
    const int nValid = (nv > 128) ? 128 : (int)nv;

    const int si = tid >> 1;
    const int sj = (tid & 1) * 16;
    const unsigned ldsA = (unsigned)(uintptr_t)(&Af[si * ASTRIDE + sj]);
    const unsigned ldsB = (unsigned)(uintptr_t)(&Bf[si * ASTRIDE + sj]);
    const float* aBase = ws_t1;
    const float* bBase = Wv2 + nbase * BH;
    const unsigned rowOff = (unsigned)((si * BH + sj) * 4);
    const bool bRowOk = (si < nValid);

    v8f acc[4][2];
#pragma unroll
    for (int im = 0; im < 4; ++im)
#pragma unroll
        for (int in = 0; in < 2; ++in) acc[im][in] = (v8f){};

    async_stage_row16(ldsA, aBase, rowOff);
    if (bRowOk) async_stage_row16(ldsB, bBase, rowOff);
    wait_async_all();
    __syncthreads();

    v16bf af[4], bfr[2];
#pragma unroll
    for (int im = 0; im < 4; ++im) af[im] = load_afrag_f32(Af, wy * 4 + im, lane);
#pragma unroll
    for (int in = 0; in < 2; ++in) bfr[in] = load_bfrag_f32(Bf, wx * 2 + in, lane);

    const int NS = BH / 32;
    for (int ks = 0; ks < NS; ++ks) {
        __syncthreads();
        const bool more = (ks + 1 < NS);
        if (more) {
            const unsigned off = rowOff + (unsigned)((ks + 1) * 32 * 4);
            async_stage_row16(ldsA, aBase, off);
            if (bRowOk) async_stage_row16(ldsB, bBase, off);
        }
#pragma unroll
        for (int im = 0; im < 4; ++im)
#pragma unroll
            for (int in = 0; in < 2; ++in)
                acc[im][in] = __builtin_amdgcn_wmma_f32_16x16x32_bf16(
                    false, af[im], false, bfr[in], (short)0, acc[im][in],
                    false, false);
        if (more) {
            wait_async_all();
            __syncthreads();
#pragma unroll
            for (int im = 0; im < 4; ++im) af[im] = load_afrag_f32(Af, wy * 4 + im, lane);
#pragma unroll
            for (int in = 0; in < 2; ++in) bfr[in] = load_bfrag_f32(Bf, wx * 2 + in, lane);
        }
    }

#pragma unroll
    for (int im = 0; im < 4; ++im) {
        const int mt = wy * 4 + im;
#pragma unroll
        for (int in = 0; in < 2; ++in) {
            const int n = (int)nbase + (wx * 2 + in) * 16 + (lane & 15);
            if (n < BV) {
                const float bb = bv2[n];
#pragma unroll
                for (int r = 0; r < 8; ++r) {
                    const int m = mt * 16 + ((lane & 16) ? r + 8 : r);
                    outF[(long)m * Vext + n] = acc[im][in][r] + bb;
                }
            }
        }
    }
}

// ---------------------------------------------------------------------------
// K6: in-place vocab softmax * p_gen, zero tail, scatter-add (1-p_gen)*attn
// ---------------------------------------------------------------------------
__global__ __launch_bounds__(256)
void k6_final(float* outF, const float* out_pg, const float* attn,
              const int* ext_ids, int Vext)
{
    const int b = blockIdx.x, t = threadIdx.x;
    float* row = outF + (long)b * Vext;
    __shared__ float red[256];

    float m = -__builtin_inff();
    for (int v = t; v < BV; v += 256) m = fmaxf(m, row[v]);
    red[t] = m; __syncthreads();
    for (int s = 128; s > 0; s >>= 1) {
        if (t < s) red[t] = fmaxf(red[t], red[t + s]);
        __syncthreads();
    }
    m = red[0]; __syncthreads();

    float sum = 0.0f;
    for (int v = t; v < BV; v += 256) sum += __expf(row[v] - m);
    red[t] = sum; __syncthreads();
    for (int s = 128; s > 0; s >>= 1) {
        if (t < s) red[t] += red[t + s];
        __syncthreads();
    }
    sum = red[0];

    const float pg = out_pg[b];
    const float scale = pg / sum;
    for (int v = t; v < BV; v += 256) row[v] = __expf(row[v] - m) * scale;
    for (int v = BV + t; v < Vext; v += 256) row[v] = 0.0f;
    __syncthreads();

    const float wad = 1.0f - pg;
    for (int s = t; s < BS; s += 256) {
        const int idx = ext_ids[(long)b * BS + s];
        atomicAdd(&row[idx], wad * attn[(long)b * BS + s]);
    }
}

extern "C" void kernel_launch(void* const* d_in, const int* in_sizes, int n_in,
                              void* d_out, int out_size, void* d_ws, size_t ws_size,
                              hipStream_t stream)
{
    (void)in_sizes; (void)n_in; (void)ws_size;

    const int*   prev_target = (const int*)  d_in[0];
    const float* h0          = (const float*)d_in[1];
    const float* c0          = (const float*)d_in[2];
    const float* enc         = (const float*)d_in[3];
    const int*   ext_ids     = (const int*)  d_in[4];
    const float* prev_ctx    = (const float*)d_in[6];
    const float* prev_cov    = (const float*)d_in[7];
    const float* embedding   = (const float*)d_in[8];
    const float* Wxc         = (const float*)d_in[9];
    const float* bxc         = (const float*)d_in[10];
    const float* W_ih        = (const float*)d_in[11];
    const float* W_hh        = (const float*)d_in[12];
    const float* b_ih        = (const float*)d_in[13];
    const float* b_hh        = (const float*)d_in[14];
    const float* Wh          = (const float*)d_in[15];
    const float* bh          = (const float*)d_in[16];
    const float* Ws          = (const float*)d_in[17];
    const float* bs          = (const float*)d_in[18];
    const float* Wc          = (const float*)d_in[19];
    const float* bc          = (const float*)d_in[20];
    const float* vvec        = (const float*)d_in[21];
    const float* Wv1         = (const float*)d_in[22];
    const float* bv1         = (const float*)d_in[23];
    const float* Wv2         = (const float*)d_in[24];
    const float* bv2         = (const float*)d_in[25];
    const float* Wpg         = (const float*)d_in[26];
    const float* bpg         = (const float*)d_in[27];

    const int tailElems = BB*BH + BB*BH + BB*H2 + BB*BS + BB + BB*BS;
    const int Vext = (out_size - tailElems) / BB;   // V + oov (= 50050)
    float* out       = (float*)d_out;
    float* out_final = out;
    float* out_h     = out + (long)BB * Vext;
    float* out_c     = out_h   + (long)BB * BH;
    float* out_ctx   = out_c   + (long)BB * BH;
    float* out_attn  = out_ctx + (long)BB * H2;
    float* out_pg    = out_attn + (long)BB * BS;
    float* out_cov   = out_pg   + BB;

    float* ws       = (float*)d_ws;
    float* ws_x     = ws;
    float* ws_shat  = ws_x     + (long)BB * BE;
    float* ws_sbias = ws_shat  + (long)BB * H2;
    float* ws_e     = ws_sbias + (long)BB * H2;
    float* ws_t1    = ws_e     + (long)BB * BS;

    kz_zero<<<(BB * BS + 255) / 256, 256, 0, stream>>>(ws_e, BB * BS);

    k1_lstm<<<BB, 256, 0, stream>>>(prev_target, h0, c0, prev_ctx, embedding,
                                    Wxc, bxc, W_ih, W_hh, b_ih, b_hh,
                                    Ws, bs, bh, bc,
                                    out_h, out_c, ws_x, ws_shat, ws_sbias);

    k2_attn_gemm<<<dim3(H2 / 128, (BB * BS) / 128), 256, 0, stream>>>(
        enc, Wh, ws_sbias, prev_cov, Wc, vvec, ws_e);

    k3_softmax_e<<<BB, 512, 0, stream>>>(ws_e, prev_cov, out_attn, out_cov);

    k4_context<<<dim3(H2 / 256, BB), 256, 0, stream>>>(out_attn, enc, out_ctx);

    k5a_t1<<<BB, 256, 0, stream>>>(out_h, out_ctx, Wv1, bv1, ws_t1);

    k5c_pgen<<<1, BB, 0, stream>>>(ws_shat, out_ctx, ws_x, Wpg, bpg, out_pg);

    const int nBlocks5b = (BV + 127) / 128;   // 391
    k5b_vocab_gemm<<<nBlocks5b, 256, 0, stream>>>(ws_t1, Wv2, bv2,
                                                  out_final, Vext);

    k6_final<<<BB, 256, 0, stream>>>(out_final, out_pg, out_attn, ext_ids, Vext);
}
